// MultiHeadAttention_51290499449312
// MI455X (gfx1250) — matrix operand order repaired and checked by reading, not compiled, not run on hardware
//
#include <hip/hip_runtime.h>
#include <hip/hip_bf16.h>

typedef __attribute__((ext_vector_type(16))) __bf16 v16bf;
typedef __attribute__((ext_vector_type(8)))  __bf16 v8bf;
typedef __attribute__((ext_vector_type(8)))  float  v8f;

__device__ __forceinline__ v8f wmma_bf16(v16bf a, v16bf b, v8f c) {
  // 8 args: (neg_a, A, neg_b, B, c_mod, C, reuse_a, reuse_b)
  return __builtin_amdgcn_wmma_f32_16x16x32_bf16(false, a, false, b, (short)0, c,
                                                 false, false);
}

__device__ __forceinline__ v16bf cat16(v8bf lo, v8bf hi) {
  v16bf r;
#pragma unroll
  for (int i = 0; i < 8; ++i) { r[i] = lo[i]; r[8 + i] = hi[i]; }
  return r;
}

// ---------------------------------------------------------------------------
// Weight transpose + f32->bf16 convert: WT[n*K + k] = (bf16) W[k*N + n]
// ---------------------------------------------------------------------------
__global__ __launch_bounds__(256) void transpose_w_kernel(
    const float* __restrict__ W, __bf16* __restrict__ WT, int K, int N) {
  const int idx = blockIdx.x * 256 + threadIdx.x;
  if (idx < K * N) {
    const int k = idx / N;
    const int n = idx - k * N;
    WT[(size_t)n * K + k] = (__bf16)W[idx];
  }
}

// ---------------------------------------------------------------------------
// Y[M,N] = A[M,K] * BT[N,K]^T  (+bias).  One wave -> 16x64 output tile.
// A is f32 (converted on the fly) or bf16. Output: bf16 (scaled by oscale),
// bf16-transposed (Vt[b, col, t], T=2048 hardcoded), or f32 (+bias).
// ---------------------------------------------------------------------------
template <bool A_F32, bool OUT_F32, bool TRANS_OUT>
__global__ __launch_bounds__(256) void gemm_bt_kernel(
    const void* __restrict__ Ap, const __bf16* __restrict__ BT,
    const float* __restrict__ bias, void* __restrict__ Yp,
    int M, int N, int K, float oscale) {
  const int lane = threadIdx.x & 31;
  const int gw = blockIdx.x * 8 + (threadIdx.x >> 5);
  const int tn_count = N >> 6;
  const int tm = gw / tn_count;
  const int tn = gw - tm * tn_count;
  const int nl = lane & 15;
  const int hi = lane >> 4;
  const int arow = tm * 16 + nl;

  v8f acc0 = {}, acc1 = {}, acc2 = {}, acc3 = {};

  for (int k0 = 0; k0 < K; k0 += 32) {
    v16bf a;
    if (A_F32) {
      const float* ap = (const float*)Ap + (size_t)arow * K + k0 + hi * 16;
      __builtin_prefetch(ap + 32, 0, 3);       // next k-chunk of A (WGP scope)
#pragma unroll
      for (int j = 0; j < 8; ++j) {
        a[j]     = (__bf16)ap[j];        // K = k0 + hi*16 + j
        a[8 + j] = (__bf16)ap[8 + j];    // K = k0 + hi*16 + 8 + j
      }
    } else {
      const __bf16* ap = (const __bf16*)Ap + (size_t)arow * K + k0 + hi * 16;
      __builtin_prefetch(ap + 32, 0, 3);
      a = cat16(*(const v8bf*)ap, *(const v8bf*)(ap + 8));
    }
    // B fragment: lane holds column (tn*64 + nt*16 + nl), rows k0+hi*16+0..15
    const __bf16* bp = BT + (size_t)(tn * 64 + nl) * K + k0 + hi * 16;
    __builtin_prefetch(bp + 32, 0, 3);         // next k-chunk of B panel
    acc0 = wmma_bf16(a, *(const v16bf*)bp, acc0); bp += (size_t)16 * K;
    acc1 = wmma_bf16(a, *(const v16bf*)bp, acc1); bp += (size_t)16 * K;
    acc2 = wmma_bf16(a, *(const v16bf*)bp, acc2); bp += (size_t)16 * K;
    acc3 = wmma_bf16(a, *(const v16bf*)bp, acc3);
  }

#pragma unroll
  for (int r = 0; r < 8; ++r) {
    const int orow = tm * 16 + hi * 8 + r;
    const int oc = tn * 64 + nl;
    if (OUT_F32) {
      float* Y = (float*)Yp;
      const size_t base = (size_t)orow * N + oc;
      Y[base +  0] = acc0[r] + bias[oc +  0];
      Y[base + 16] = acc1[r] + bias[oc + 16];
      Y[base + 32] = acc2[r] + bias[oc + 32];
      Y[base + 48] = acc3[r] + bias[oc + 48];
    } else if (TRANS_OUT) {
      __bf16* Y = (__bf16*)Yp;
      const int bb = orow >> 11, t = orow & 2047;
      const size_t base = ((size_t)bb * N + oc) * 2048 + t;
      Y[base + (size_t) 0 * 2048] = (__bf16)(acc0[r] * oscale);
      Y[base + (size_t)16 * 2048] = (__bf16)(acc1[r] * oscale);
      Y[base + (size_t)32 * 2048] = (__bf16)(acc2[r] * oscale);
      Y[base + (size_t)48 * 2048] = (__bf16)(acc3[r] * oscale);
    } else {
      __bf16* Y = (__bf16*)Yp;
      const size_t base = (size_t)orow * N + oc;
      Y[base +  0] = (__bf16)(acc0[r] * oscale);
      Y[base + 16] = (__bf16)(acc1[r] * oscale);
      Y[base + 32] = (__bf16)(acc2[r] * oscale);
      Y[base + 48] = (__bf16)(acc3[r] * oscale);
    }
  }
}

// ---------------------------------------------------------------------------
// Flash attention: one wave per (b, h, 16-query tile). Online softmax.
// Qb (pre-scaled by 1/sqrt(64)), Kb: bf16 [B*T, 1024] row-major.
// Vt: bf16 [B, 1024, T] (feature-major so PV B-fragments are contiguous).
// Ob: bf16 [B*T, 1024].
// ---------------------------------------------------------------------------
__global__ __launch_bounds__(256) void attn_kernel(
    const __bf16* __restrict__ Qb, const __bf16* __restrict__ Kb,
    const __bf16* __restrict__ Vt, __bf16* __restrict__ Ob) {
  __shared__ float pbuf[8][16][32];   // per-wave P-tile bounce buffer (16 KB)
  const int wid  = threadIdx.x >> 5;
  const int lane = threadIdx.x & 31;
  const int gw   = blockIdx.x * 8 + wid;
  const int qt = gw & 127;          // T/16 = 128 query tiles
  const int h  = (gw >> 7) & 15;
  const int b  = gw >> 11;
  const int nl = lane & 15;
  const int hi = lane >> 4;
  const int q0 = qt * 16;

  // Q A-fragments (held for whole kernel): rows q0+nl, feature dim 0..63
  const __bf16* qp = Qb + (size_t)(b * 2048 + q0 + nl) * 1024 + h * 64 + hi * 16;
  const v16bf aq0 = cat16(*(const v8bf*)qp,        *(const v8bf*)(qp + 8));
  const v16bf aq1 = cat16(*(const v8bf*)(qp + 32), *(const v8bf*)(qp + 40));

  const __bf16* kbbase = Kb + (size_t)b * 2048 * 1024 + h * 64 + hi * 16;
  const __bf16* vtbase = Vt + ((size_t)b * 1024 + h * 64 + nl) * 2048 + hi * 16;

  v8f o0 = {}, o1 = {}, o2 = {}, o3 = {};
  float mrow[8], lrow[8];
#pragma unroll
  for (int r = 0; r < 8; ++r) { mrow[r] = -__builtin_inff(); lrow[r] = 0.0f; }

  float (*pb)[32] = pbuf[wid];

  for (int kb = 0; kb <= q0 + 15; kb += 32) {
    // ---- scores: S(16x32) = Q(16x64) * K^T  (Q pre-scaled by 1/sqrt(d)) ----
    v8f s0 = {}, s1 = {};
    {
      const __bf16* k0p = kbbase + (size_t)(kb + nl) * 1024;
      __builtin_prefetch(k0p + (size_t)32 * 1024, 0, 3);  // next chunk's K rows
      __builtin_prefetch(k0p + (size_t)48 * 1024, 0, 3);
      s0 = wmma_bf16(aq0, *(const v16bf*)k0p, s0);
      s0 = wmma_bf16(aq1, *(const v16bf*)(k0p + 32), s0);
      const __bf16* k1p = k0p + (size_t)16 * 1024;
      s1 = wmma_bf16(aq0, *(const v16bf*)k1p, s1);
      s1 = wmma_bf16(aq1, *(const v16bf*)(k1p + 32), s1);
    }
    // ---- causal mask + online softmax ----
    const float NEG = -__builtin_inff();
#pragma unroll
    for (int r = 0; r < 8; ++r) {
      const int q = q0 + hi * 8 + r;                 // this row's query index
      s0[r] = (kb + nl      > q) ? NEG : s0[r];
      s1[r] = (kb + 16 + nl > q) ? NEG : s1[r];
    }
#pragma unroll
    for (int r = 0; r < 8; ++r) {
      float t = fmaxf(s0[r], s1[r]);                 // row max over 16 lanes
      t = fmaxf(t, __shfl_xor(t, 1));
      t = fmaxf(t, __shfl_xor(t, 2));
      t = fmaxf(t, __shfl_xor(t, 4));
      t = fmaxf(t, __shfl_xor(t, 8));
      const float mn = fmaxf(mrow[r], t);
      const float alpha = __expf(mrow[r] - mn);
      mrow[r] = mn;
      const float p0 = __expf(s0[r] - mn);
      const float p1 = __expf(s1[r] - mn);
      float rs = p0 + p1;                            // row sum over 16 lanes
      rs += __shfl_xor(rs, 1);
      rs += __shfl_xor(rs, 2);
      rs += __shfl_xor(rs, 4);
      rs += __shfl_xor(rs, 8);
      lrow[r] = lrow[r] * alpha + rs;
      pb[hi * 8 + r][nl]      = p0;                  // stash P tile (C layout)
      pb[hi * 8 + r][16 + nl] = p1;
      o0[r] *= alpha; o1[r] *= alpha; o2[r] *= alpha; o3[r] *= alpha;
    }
    // ---- C-layout -> A-layout conversion through LDS (intra-wave, in-order)
    v16bf pa;
#pragma unroll
    for (int j = 0; j < 8; ++j) {
      pa[j]     = (__bf16)pb[nl][hi * 16 + j];
      pa[8 + j] = (__bf16)pb[nl][hi * 16 + 8 + j];
    }
    // ---- O(16x64) += P(16x32) * V(32x64) ----
    const __bf16* vp = vtbase + kb;
    __builtin_prefetch(vp + 32, 0, 3);               // next chunk of V rows
    o0 = wmma_bf16(pa, *(const v16bf*)vp, o0);
    o1 = wmma_bf16(pa, *(const v16bf*)(vp + (size_t)16 * 2048), o1);
    o2 = wmma_bf16(pa, *(const v16bf*)(vp + (size_t)32 * 2048), o2);
    o3 = wmma_bf16(pa, *(const v16bf*)(vp + (size_t)48 * 2048), o3);
  }

  // ---- normalize and store O (bf16) ----
#pragma unroll
  for (int r = 0; r < 8; ++r) {
    const float inv = 1.0f / lrow[r];
    __bf16* op = Ob + (size_t)(b * 2048 + q0 + hi * 8 + r) * 1024 + h * 64 + nl;
    op[0]  = (__bf16)(o0[r] * inv);
    op[16] = (__bf16)(o1[r] * inv);
    op[32] = (__bf16)(o2[r] * inv);
    op[48] = (__bf16)(o3[r] * inv);
  }
}

// ---------------------------------------------------------------------------
extern "C" void kernel_launch(void* const* d_in, const int* in_sizes, int n_in,
                              void* d_out, int out_size, void* d_ws, size_t ws_size,
                              hipStream_t stream) {
  const float* x  = (const float*)d_in[0];
  const float* Wq = (const float*)d_in[1];
  const float* Wk = (const float*)d_in[2];
  const float* Wv = (const float*)d_in[3];
  const float* Wc = (const float*)d_in[4];
  const float* bc = (const float*)d_in[5];
  float* out = (float*)d_out;

  constexpr int Bn = 4, T = 2048, C = 1024;
  constexpr int M = Bn * T;                               // 8192
  constexpr size_t wbytes = (size_t)C * C * sizeof(__bf16);  // 2 MB
  constexpr size_t abytes = (size_t)M * C * sizeof(__bf16);  // 16 MB

  char* p = (char*)d_ws;
  __bf16* WqT = (__bf16*)p; p += wbytes;
  __bf16* WkT = (__bf16*)p; p += wbytes;
  __bf16* WvT = (__bf16*)p; p += wbytes;
  __bf16* WcT = (__bf16*)p; p += wbytes;
  __bf16* Qb  = (__bf16*)p; p += abytes;
  __bf16* Kb  = (__bf16*)p; p += abytes;
  __bf16* Vt  = (__bf16*)p; p += abytes;
  __bf16* Ob  = (__bf16*)p; p += abytes;   // total 72 MB

  const int tblocks = (C * C + 255) / 256;                // 4096
  transpose_w_kernel<<<tblocks, 256, 0, stream>>>(Wq, WqT, C, C);
  transpose_w_kernel<<<tblocks, 256, 0, stream>>>(Wk, WkT, C, C);
  transpose_w_kernel<<<tblocks, 256, 0, stream>>>(Wv, WvT, C, C);
  transpose_w_kernel<<<tblocks, 256, 0, stream>>>(Wc, WcT, C, C);

  const int gblocks = (M / 16) * (C / 64) / 8;            // 1024 blocks, 8 waves each
  // Q is pre-scaled by 1/sqrt(HEAD_DIM) = 0.125 (exact power of 2: no extra
  // rounding vs scaling the scores, and it removes muls from the attn hot loop)
  gemm_bt_kernel<true,  false, false><<<gblocks, 256, 0, stream>>>(x, WqT, nullptr, Qb, M, C, C, 0.125f);
  gemm_bt_kernel<true,  false, false><<<gblocks, 256, 0, stream>>>(x, WkT, nullptr, Kb, M, C, C, 1.0f);
  gemm_bt_kernel<true,  false, true ><<<gblocks, 256, 0, stream>>>(x, WvT, nullptr, Vt, M, C, C, 1.0f);

  // B*H*(T/16) = 8192 waves -> 1024 blocks
  attn_kernel<<<1024, 256, 0, stream>>>(Qb, Kb, Vt, Ob);

  gemm_bt_kernel<false, true,  false><<<gblocks, 256, 0, stream>>>(Ob, WcT, bc, out, M, C, C, 1.0f);

  (void)in_sizes; (void)n_in; (void)out_size; (void)ws_size;
}
